// GRULayer_67903432950127
// MI455X (gfx1250) — matrix operand order repaired and checked by reading, not compiled, not run on hardware
//
#include <hip/hip_runtime.h>
#include <hip/hip_bf16.h>
#include <math.h>

// GRU: B=64, T=512, I=512, H=1024.
// Step t computes [inp_t | h_prev] (64 x 1536) @ [W_in; W_hid] (1536 x 3072)
// with v_wmma_f32_16x16x32_bf16. Weights are pre-swizzled once into the WMMA
// B-operand per-lane layout (bf16, 9.4 MB -> resident in the 192 MB L2 for
// all 512 steps). Each 16x16 output tile is computed by a 4-wave workgroup
// using split-K (12 K-chunks per wave) + an LDS reduction => 1024 waves/step
// to hide memory latency on the sequential critical path. The input/hidden
// K-boundary is handled by two straight-line loops (no per-iteration branch).

typedef __attribute__((ext_vector_type(16))) __bf16 v16bf;
typedef __attribute__((ext_vector_type(8)))  float  v8f;

#define Bsz  64
#define Tsz  512
#define Isz  512
#define Hsz  1024
#define G3H  3072
#define KTOT 1536
#define KCH  48     // K chunks of 32 (16 input chunks + 32 hidden chunks)
#define KCH_IN 16   // chunks covering the input part (K < 512)
#define NT   192    // 16-wide column tiles across 3H
#define KSPLIT 4    // waves per output tile (split-K)
#define KPW  (KCH / KSPLIT)   // 12 K-chunks per wave

// ---------------------------------------------------------------------------
// One-time weight swizzle: f32 row-major [1536,3072] (W_in stacked on W_hid)
// -> bf16 chunks so each lane of a wave loads its 16 K-values as one
// contiguous 32B run:  chunk(ntile, ktile)[lane][0..15] with
//   column n = ntile*16 + lane%16
//   K       = ktile*32 + (lane>=16 ? 16 : 0) + j      (j ascending)
// ---------------------------------------------------------------------------
__global__ __launch_bounds__(256) void gru_swizzle_w(
    const float* __restrict__ Win, const float* __restrict__ Whid,
    __bf16* __restrict__ Wb) {
  int gid = blockIdx.x * blockDim.x + threadIdx.x;     // (tile, lane)
  const int total = KCH * NT * 32;
  if (gid >= total) return;
  int lane = gid & 31;
  int tile = gid >> 5;
  int kt = tile % KCH;
  int nt = tile / KCH;
  int n     = nt * 16 + (lane & 15);
  int kbase = kt * 32 + ((lane >> 4) ? 16 : 0);
  __bf16* dst = Wb + (((size_t)nt * KCH + kt) * 32 + lane) * 16;
#pragma unroll
  for (int j = 0; j < 16; ++j) {
    int k = kbase + j;
    float v = (k < Isz) ? Win[(size_t)k * G3H + n]
                        : Whid[(size_t)(k - Isz) * G3H + n];
    dst[j] = (__bf16)v;
  }
}

// Broadcast hid_init [1,H] to h0 [B,H].
__global__ __launch_bounds__(256) void gru_init_h(
    const float* __restrict__ hid_init, float* __restrict__ h) {
  int gid = blockIdx.x * blockDim.x + threadIdx.x;     // B*H threads
  h[gid] = hid_init[gid & (Hsz - 1)];
}

// Straight-line GEMM over nchunks K-chunks within ONE source region.
// src: this lane's A row, pre-offset so src[0] is the first K of this lane's
// slice. accc receives the gate-c partial (accxc for the input region,
// acchc for the hidden region).
__device__ __forceinline__ void gru_gemm_range(
    int nchunks, const float* __restrict__ src,
    const v16bf*& pR, const v16bf*& pU, const v16bf*& pC,
    v8f& accr, v8f& accu, v8f& accc) {
  const int STEPV = 32;                                // 32 v16bf = 1024 B
  for (int kc = 0; kc < nchunks; ++kc) {
    // Unconditionally prefetch the next K-chunk's B tiles (L2-resident).
    // One-past-range prefetches hit the next wave's shared chunks / the h
    // scratch region of the same workspace allocation -- harmless.
    __builtin_prefetch((const char*)pR + 1024, 0, 1);
    __builtin_prefetch((const char*)pU + 1024, 0, 1);
    __builtin_prefetch((const char*)pC + 1024, 0, 1);

    // A layout used here (contiguous K, matching Wb): lanes 0-15 hold K = +0..15,
    //                  lanes 16-31 hold K = +16..31 (src pre-offset).
    v16bf a;
#pragma unroll
    for (int j = 0; j < 8; ++j) a[j]     = (__bf16)src[j];
#pragma unroll
    for (int j = 0; j < 8; ++j) a[8 + j] = (__bf16)src[8 + j];
    src += 32;

    v16bf bR = *pR, bU = *pU, bC = *pC;
    pR += STEPV; pU += STEPV; pC += STEPV;

    accr = __builtin_amdgcn_wmma_f32_16x16x32_bf16(false, a, false, bR,
                                                   (short)0, accr, false, false);
    accu = __builtin_amdgcn_wmma_f32_16x16x32_bf16(false, a, false, bU,
                                                   (short)0, accu, false, false);
    accc = __builtin_amdgcn_wmma_f32_16x16x32_bf16(false, a, false, bC,
                                                   (short)0, accc, false, false);
  }
}

// ---------------------------------------------------------------------------
// One recurrent step. Grid: 256 blocks (one per 16x16 output tile) of 128
// threads (4 waves, split-K). Wave w covers K-chunks [w*12, w*12+12), split
// at the input/hidden boundary (chunk 16) into two branch-free loops.
// Gate c keeps input-part and hidden-part partial sums separate
// (c = tanh(x_c + r*h_c)); the LDS reduction sums all four accumulators
// across waves, then wave 0 runs the fused nonlinearity + h-update tail.
// ---------------------------------------------------------------------------
__global__ __launch_bounds__(128) void gru_step(
    const float* __restrict__ inp, const __bf16* __restrict__ Wb,
    const float* __restrict__ bias, const float* __restrict__ hprev,
    float* __restrict__ hout, int t) {
  int widx  = threadIdx.x >> 5;                        // 0..3 (K-split slice)
  int lane  = threadIdx.x & 31;
  int tile  = blockIdx.x;                              // 0..255
  int mt    = tile >> 6;                               // 0..3   (batch tile)
  int nt    = tile & 63;                               // 0..63  (hidden tile)
  int hi    = lane >> 4;
  int lan15 = lane & 15;
  int mbase = mt * 16;
  int lofs  = hi ? 16 : 0;                             // per-lane K offset

  v8f accr = {}, accu = {}, accxc = {}, acchc = {};

  int kc0 = widx * KPW, kc1 = kc0 + KPW;
  int nin  = min(kc1, KCH_IN) - min(kc0, KCH_IN);      // chunks in input part
  int nhid = (kc1 - kc0) - nin;                        // chunks in hidden part

  // Swizzled B chunk pointers: consecutive K-chunks are +32*16 bf16 = 1024 B.
  const v16bf* pR = (const v16bf*)(Wb + (((size_t)(0 * 64 + nt) * KCH + kc0) * 32 + lane) * 16);
  const v16bf* pU = (const v16bf*)(Wb + (((size_t)(1 * 64 + nt) * KCH + kc0) * 32 + lane) * 16);
  const v16bf* pC = (const v16bf*)(Wb + (((size_t)(2 * 64 + nt) * KCH + kc0) * 32 + lane) * 16);

  if (nin > 0) {
    const float* src = inp + ((size_t)(mbase + lan15) * Tsz + t) * Isz
                           + kc0 * 32 + lofs;
    gru_gemm_range(nin, src, pR, pU, pC, accr, accu, accxc);
  }
  if (nhid > 0) {
    int kcH = max(kc0, KCH_IN);
    const float* src = hprev + (size_t)(mbase + lan15) * Hsz
                             + (kcH - KCH_IN) * 32 + lofs;
    gru_gemm_range(nhid, src, pR, pU, pC, accr, accu, acchc);
  }

  // Split-K reduction through LDS: waves 1..3 store 4 accumulators each.
  __shared__ __align__(16) float red[KSPLIT - 1][4][32][8];
  if (widx > 0) {
    *(v8f*)&red[widx - 1][0][lane][0] = accr;
    *(v8f*)&red[widx - 1][1][lane][0] = accu;
    *(v8f*)&red[widx - 1][2][lane][0] = accxc;
    *(v8f*)&red[widx - 1][3][lane][0] = acchc;
  }
  __syncthreads();
  if (widx != 0) return;

#pragma unroll
  for (int w = 0; w < KSPLIT - 1; ++w) {
    accr  += *(const v8f*)&red[w][0][lane][0];
    accu  += *(const v8f*)&red[w][1][lane][0];
    accxc += *(const v8f*)&red[w][2][lane][0];
    acchc += *(const v8f*)&red[w][3][lane][0];
  }

  // Fused gate nonlinearities + state update, using the C/D register layout:
  // acc element v <-> (M = v + 8*hi, N = lane%16).
  int j = nt * 16 + lan15;
  float br = bias[j], bu = bias[Hsz + j], bc = bias[2 * Hsz + j];
#pragma unroll
  for (int v = 0; v < 8; ++v) {
    int brow = mbase + v + (hi ? 8 : 0);
    float r  = 1.0f / (1.0f + __expf(-(accr[v] + br)));
    float u  = 1.0f / (1.0f + __expf(-(accu[v] + bu)));
    float c  = tanhf(accxc[v] + bc + r * acchc[v]);
    float hp = hprev[(size_t)brow * Hsz + j];
    hout[(size_t)brow * Hsz + j] = (1.0f - u) * hp + u * c;
  }
}

extern "C" void kernel_launch(void* const* d_in, const int* in_sizes, int n_in,
                              void* d_out, int out_size, void* d_ws, size_t ws_size,
                              hipStream_t stream) {
  const float* inp      = (const float*)d_in[0];  // [B,T,I]
  const float* Win      = (const float*)d_in[1];  // [I,3H]
  const float* Whid     = (const float*)d_in[2];  // [H,3H]
  const float* bias     = (const float*)d_in[3];  // [3H]
  const float* hid_init = (const float*)d_in[4];  // [1,H]
  float* out = (float*)d_out;                     // [B,H]

  // Workspace: swizzled bf16 weights (9.4 MB) + double-buffered h (2 x 256 KB).
  __bf16* Wb = (__bf16*)d_ws;
  size_t wb_bytes = (size_t)KTOT * G3H * sizeof(__bf16);
  float* h0 = (float*)((char*)d_ws + wb_bytes);
  float* h1 = h0 + Bsz * Hsz;

  {
    int total = KCH * NT * 32;
    gru_swizzle_w<<<(total + 255) / 256, 256, 0, stream>>>(Win, Whid, Wb);
  }
  gru_init_h<<<(Bsz * Hsz) / 256, 256, 0, stream>>>(hid_init, h0);

  for (int t = 0; t < Tsz; ++t) {
    const float* hp = (t & 1) ? h1 : h0;
    float*       hn = (t == Tsz - 1) ? out : ((t & 1) ? h0 : h1);
    gru_step<<<256, 128, 0, stream>>>(inp, Wb, bias, hp, hn, t);
  }
}
